// CodeGraphModel_89764816487072
// MI455X (gfx1250) — matrix operand order repaired and checked by reading, not compiled, not run on hardware
//
#include <hip/hip_runtime.h>

typedef __bf16 v16bf __attribute__((ext_vector_type(16)));
typedef float v8f __attribute__((ext_vector_type(8)));
typedef unsigned int uv4 __attribute__((ext_vector_type(4)));
typedef float fv4 __attribute__((ext_vector_type(4)));
typedef unsigned short ushort_t;
typedef unsigned int uint_t;

#define AGG_TPW 8   // edge tiles (16 edges each) per wave in the aggregate kernel

// ---------- helpers ----------

__device__ __forceinline__ ushort_t bf_bits(float f) {
  __bf16 b = (__bf16)f;                       // hardware RNE f32->bf16 (v_cvt_pk_bf16_f32)
  return __builtin_bit_cast(ushort_t, b);
}

// order-preserving float -> uint key (for branchless atomic max)
__device__ __forceinline__ uint_t enc_key(float f) {
  uint_t u = __float_as_uint(f);
  return ((int)u >= 0) ? (u | 0x80000000u) : ~u;
}
// key -> float, with -inf (no-edge nodes) cleaned to 0
__device__ __forceinline__ float dec_key(uint_t k) {
  uint_t u = (k & 0x80000000u) ? (k & 0x7FFFFFFFu) : ~k;
  float f = __uint_as_float(u);
  return (f >= -1e30f) ? f : 0.0f;
}

// 16 bf16 elements stored contiguously (32 bytes) -> WMMA fragment
__device__ __forceinline__ v16bf frag_bf16(const ushort_t* p) {
  union { uv4 q[2]; v16bf v; } u;
  u.q[0] = *(const uv4*)(p);
  u.q[1] = *(const uv4*)(p + 8);
  return u.v;
}

__device__ __forceinline__ v16bf cvt16(const float* e) {
  v16bf v;
#pragma unroll
  for (int i = 0; i < 16; ++i) v[i] = (__bf16)e[i];
  return v;
}

// A-fragment from an f32 row: elements p[0..15]
__device__ __forceinline__ v16bf fragA_f32(const float* p) {
  fv4 a0 = *(const fv4*)(p);
  fv4 a1 = *(const fv4*)(p + 4);
  fv4 b0 = *(const fv4*)(p + 8);
  fv4 b1 = *(const fv4*)(p + 12);
  float e[16] = {a0.x, a0.y, a0.z, a0.w, a1.x, a1.y, a1.z, a1.w,
                 b0.x, b0.y, b0.z, b0.w, b1.x, b1.y, b1.z, b1.w};
  return cvt16(e);
}

// A-fragment from an ordered-key row (the segment-max accumulator)
__device__ __forceinline__ v16bf fragA_key(const uint_t* p) {
  uv4 a0 = *(const uv4*)(p);
  uv4 a1 = *(const uv4*)(p + 4);
  uv4 b0 = *(const uv4*)(p + 8);
  uv4 b1 = *(const uv4*)(p + 12);
  uint_t k[16] = {a0.x, a0.y, a0.z, a0.w, a1.x, a1.y, a1.z, a1.w,
                  b0.x, b0.y, b0.z, b0.w, b1.x, b1.y, b1.z, b1.w};
  float e[16];
#pragma unroll
  for (int i = 0; i < 16; ++i) e[i] = dec_key(k[i]);
  return cvt16(e);
}

__device__ __forceinline__ v8f wmma_bf16(v16bf a, v16bf b, v8f c) {
  return __builtin_amdgcn_wmma_f32_16x16x32_bf16(false, a, false, b, (short)0, c,
                                                 false, false);
}

__device__ __forceinline__ float sigmoidf_(float x) {
  return 1.0f / (1.0f + __expf(-x));
}

// ---------- weight pre-swizzle: row-major f32 [batch,K,NC] -> bf16 B-fragments ----------
// Fragment (b,kt,nt): lane = 16*(k16>=16) + n_local, 16 bf16 per lane contiguous.
__global__ void swizzle_w_kernel(const float* __restrict__ W, ushort_t* __restrict__ dst,
                                 int K, int NC, int total) {
  int i = blockIdx.x * blockDim.x + threadIdx.x;
  if (i >= total) return;
  int per = K * NC;
  int b = i / per, rem = i - b * per;
  int k = rem / NC, n = rem - k * NC;
  int kt = k >> 5, kl = k & 31;
  int nt = n >> 4, nl = n & 15;
  int lane = ((kl >> 4) << 4) | nl;
  int idx = kl & 15;
  int NT = NC >> 4, KT = K >> 5;
  size_t o = (((size_t)b * KT * NT) + (size_t)kt * NT + nt) * 512 + (size_t)lane * 16 + idx;
  dst[o] = bf_bits(W[i]);
}

// ---------- embedding mean-pool ----------
__global__ void embed_kernel(const int* __restrict__ toks, const float* __restrict__ table,
                             float* __restrict__ emb, int N) {
  int i = blockIdx.x * blockDim.x + threadIdx.x;
  if (i >= N * 64) return;
  int nn = i >> 6, d = i & 63;
  const int* tp = toks + (size_t)nn * 5;
  float s = 0.0f;
#pragma unroll
  for (int k = 0; k < 5; ++k) s += table[(size_t)tp[k] * 64 + d];
  emb[i] = s * 0.2f;
}

// ---------- fill msg keys with key(-inf) ----------
__global__ void fill_key_kernel(uint_t* __restrict__ p, int n) {
  int i = blockIdx.x * blockDim.x + threadIdx.x;
  if (i < n) p[i] = 0x007FFFFFu;  // enc_key(-inf)
}

// ---------- h_init = emb @ proj_W (seeds h, h_bf, hinit_bf) ----------
template <bool FULL>
__device__ __forceinline__ void init_epilogue(int wid, int m, int half, int N,
                                              const v8f* acc, float* __restrict__ h,
                                              ushort_t* __restrict__ h_bf,
                                              ushort_t* __restrict__ hinit_bf) {
#pragma unroll
  for (int nt = 0; nt < 4; ++nt) {
#pragma unroll
    for (int rr = 0; rr < 8; ++rr) {
      int node = wid * 16 + rr + 8 * half;
      if (!FULL && node >= N) continue;
      size_t o = (size_t)node * 64 + nt * 16 + m;
      float v = acc[nt][rr];
      h[o] = v;
      ushort_t bb = bf_bits(v);
      h_bf[o] = bb;
      hinit_bf[o] = bb;
    }
  }
}

__global__ void init_h_kernel(const float* __restrict__ emb, const ushort_t* __restrict__ projS,
                              float* __restrict__ h, ushort_t* __restrict__ h_bf,
                              ushort_t* __restrict__ hinit_bf, int N) {
  int wid = (blockIdx.x * blockDim.x + threadIdx.x) >> 5;
  int lane = threadIdx.x & 31;
  int numTiles = (N + 15) >> 4;
  if (wid >= numTiles) return;
  int m = lane & 15, half = lane >> 4;
  int rowA = wid * 16 + m; if (rowA >= N) rowA = N - 1;
  v16bf a0 = fragA_f32(emb + (size_t)rowA * 64 + half * 16);
  v16bf a1 = fragA_f32(emb + (size_t)rowA * 64 + 32 + half * 16);
  v8f acc[4];
#pragma unroll
  for (int nt = 0; nt < 4; ++nt) {
    v8f c = {};
    c = wmma_bf16(a0, frag_bf16(projS + ((size_t)(0 * 4 + nt) * 32 + lane) * 16), c);
    c = wmma_bf16(a1, frag_bf16(projS + ((size_t)(1 * 4 + nt) * 32 + lane) * 16), c);
    acc[nt] = c;
  }
  if (wid * 16 + 16 <= N) init_epilogue<true>(wid, m, half, N, acc, h, h_bf, hinit_bf);
  else                    init_epilogue<false>(wid, m, half, N, acc, h, h_bf, hinit_bf);
}

// ---------- edge aggregation: msgKey[dst] = max over edges of key(h[src] @ W_t) ----------
template <bool FULL>
__device__ __forceinline__ void scatter_max(uint_t* __restrict__ msgK, const v8f* acc,
                                            const int* dn, int base, int half, int m, int E) {
#pragma unroll
  for (int nt = 0; nt < 4; ++nt) {
#pragma unroll
    for (int rr = 0; rr < 8; ++rr) {
      if (!FULL && (base + rr + 8 * half) >= E) continue;
      atomicMax(msgK + (size_t)dn[rr] * 64 + nt * 16 + m, enc_key(acc[nt][rr]));
    }
  }
}

// one wave = AGG_TPW tiles of 16 edges; weight fragments kept in registers across tiles
__global__ void aggregate_kernel(const int* __restrict__ adj, const ushort_t* __restrict__ h_bf,
                                 const ushort_t* __restrict__ Wswz, uint_t* __restrict__ msgK,
                                 int E, int Ttot, int Tfwd) {
  int wid = (blockIdx.x * blockDim.x + threadIdx.x) >> 5;
  int lane = threadIdx.x & 31;
  int tilesPerType = (E + 15) >> 4;
  int wavesPerType = (tilesPerType + AGG_TPW - 1) / AGG_TPW;
  if (wid >= Ttot * wavesPerType) return;
  int t = wid / wavesPerType, wv = wid - t * wavesPerType;
  int m = lane & 15, half = lane >> 4;
  const ushort_t* Wt = Wswz + (size_t)t * 4096;  // 2 kt * 4 nt * 512
  v16bf bfr[8];                                  // f = kt*4 + nt
#pragma unroll
  for (int f = 0; f < 8; ++f)
    bfr[f] = frag_bf16(Wt + ((size_t)f * 32 + lane) * 16);
  const int* adjT = adj + (size_t)(t % Tfwd) * E * 2;
  int fwd = (t < Tfwd) ? 1 : 0;                  // backward edges swap src/dst
  for (int it = 0; it < AGG_TPW; ++it) {
    int tile = wv * AGG_TPW + it;
    if (tile >= tilesPerType) break;             // wave-uniform
    int e = tile * 16 + m;
    int ec = e < E ? e : E - 1;
    const int* ap = adjT + (size_t)ec * 2;
    int src = ap[fwd ? 0 : 1];
    int dst = ap[fwd ? 1 : 0];
    v16bf a0 = frag_bf16(h_bf + (size_t)src * 64 + half * 16);
    v16bf a1 = frag_bf16(h_bf + (size_t)src * 64 + 32 + half * 16);
    v8f acc[4];
#pragma unroll
    for (int nt = 0; nt < 4; ++nt) {
      v8f c = {};
      c = wmma_bf16(a0, bfr[nt], c);
      c = wmma_bf16(a1, bfr[4 + nt], c);
      acc[nt] = c;
    }
    int dn[8];
#pragma unroll
    for (int rr = 0; rr < 8; ++rr) dn[rr] = __shfl(dst, rr + 8 * half, 32);
    int base = tile * 16;
    if (base + 16 <= E) scatter_max<true>(msgK, acc, dn, base, half, m, E);
    else                scatter_max<false>(msgK, acc, dn, base, half, m, E);
  }
}

// ---------- GRU epilogue: gate math in C-register layout ----------
template <bool FULL, bool WRITE_BF>
__device__ __forceinline__ void gru_epilogue(int wid, int m, int half, int N,
                                             const v8f* ru, const v8f* cx, const v8f* ch,
                                             const float* __restrict__ bias,
                                             const float* __restrict__ hIn,
                                             float* __restrict__ hOut,
                                             ushort_t* __restrict__ h_bf) {
#pragma unroll
  for (int nt = 0; nt < 4; ++nt) {
    int col = nt * 16 + m;
    float br = bias[col], bu = bias[64 + col], bc = bias[128 + col];
#pragma unroll
    for (int rr = 0; rr < 8; ++rr) {
      int node = wid * 16 + rr + 8 * half;
      if (!FULL && node >= N) continue;
      size_t o = (size_t)node * 64 + col;
      float r  = sigmoidf_(ru[nt][rr] + br);
      float uu = sigmoidf_(ru[nt + 4][rr] + bu);
      float hold = hIn[o];
      float cg = tanhf(cx[nt][rr] + bc + r * ch[nt][rr]);
      float hn = uu * hold + (1.0f - uu) * cg;
      hOut[o] = hn;
      if (WRITE_BF) h_bf[o] = bf_bits(hn);
    }
  }
}

// ---------- GRU layer 0: h = GRU(msg, h) ----------
__global__ void gru0_kernel(const uint_t* __restrict__ msgK, float* __restrict__ h,
                            ushort_t* __restrict__ h_bf,
                            const ushort_t* __restrict__ WxS, const ushort_t* __restrict__ WhS,
                            const float* __restrict__ bias, int N) {
  int wid = (blockIdx.x * blockDim.x + threadIdx.x) >> 5;
  int lane = threadIdx.x & 31;
  int numTiles = (N + 15) >> 4;
  if (wid >= numTiles) return;
  int m = lane & 15, half = lane >> 4;
  int rowA = wid * 16 + m; if (rowA >= N) rowA = N - 1;
  v16bf ax0 = fragA_key(msgK + (size_t)rowA * 64 + half * 16);
  v16bf ax1 = fragA_key(msgK + (size_t)rowA * 64 + 32 + half * 16);
  v16bf ah0 = frag_bf16(h_bf + (size_t)rowA * 64 + half * 16);
  v16bf ah1 = frag_bf16(h_bf + (size_t)rowA * 64 + 32 + half * 16);
  v8f ru[8], cx[4], ch[4];
#pragma unroll
  for (int nt = 0; nt < 8; ++nt) {   // r,u gates: x@Wx + h@Wh share accumulators
    v8f c = {};
    c = wmma_bf16(ax0, frag_bf16(WxS + ((size_t)(0 * 12 + nt) * 32 + lane) * 16), c);
    c = wmma_bf16(ax1, frag_bf16(WxS + ((size_t)(1 * 12 + nt) * 32 + lane) * 16), c);
    c = wmma_bf16(ah0, frag_bf16(WhS + ((size_t)(0 * 12 + nt) * 32 + lane) * 16), c);
    c = wmma_bf16(ah1, frag_bf16(WhS + ((size_t)(1 * 12 + nt) * 32 + lane) * 16), c);
    ru[nt] = c;
  }
#pragma unroll
  for (int nt = 0; nt < 4; ++nt) {   // c gate: xc and hc separate (r * hc)
    v8f c = {};
    c = wmma_bf16(ax0, frag_bf16(WxS + ((size_t)(0 * 12 + 8 + nt) * 32 + lane) * 16), c);
    c = wmma_bf16(ax1, frag_bf16(WxS + ((size_t)(1 * 12 + 8 + nt) * 32 + lane) * 16), c);
    cx[nt] = c;
    v8f d = {};
    d = wmma_bf16(ah0, frag_bf16(WhS + ((size_t)(0 * 12 + 8 + nt) * 32 + lane) * 16), d);
    d = wmma_bf16(ah1, frag_bf16(WhS + ((size_t)(1 * 12 + 8 + nt) * 32 + lane) * 16), d);
    ch[nt] = d;
  }
  if (wid * 16 + 16 <= N)
    gru_epilogue<true, true>(wid, m, half, N, ru, cx, ch, bias, h, h, h_bf);
  else
    gru_epilogue<false, true>(wid, m, half, N, ru, cx, ch, bias, h, h, h_bf);
}

// ---------- GRU layer 1: out = GRU(concat(h_init, msg), h) ----------
__global__ void gru1_kernel(const uint_t* __restrict__ msgK, const ushort_t* __restrict__ hinit_bf,
                            const float* __restrict__ h, const ushort_t* __restrict__ h_bf,
                            const ushort_t* __restrict__ WxS, const ushort_t* __restrict__ WhS,
                            const float* __restrict__ bias, float* __restrict__ outF, int N) {
  int wid = (blockIdx.x * blockDim.x + threadIdx.x) >> 5;
  int lane = threadIdx.x & 31;
  int numTiles = (N + 15) >> 4;
  if (wid >= numTiles) return;
  int m = lane & 15, half = lane >> 4;
  int rowA = wid * 16 + m; if (rowA >= N) rowA = N - 1;
  // x = concat(h_init, msg): K=128 -> kt 0,1 from h_init, kt 2,3 from msg
  v16bf ax0 = frag_bf16(hinit_bf + (size_t)rowA * 64 + half * 16);
  v16bf ax1 = frag_bf16(hinit_bf + (size_t)rowA * 64 + 32 + half * 16);
  v16bf ax2 = fragA_key(msgK + (size_t)rowA * 64 + half * 16);
  v16bf ax3 = fragA_key(msgK + (size_t)rowA * 64 + 32 + half * 16);
  v16bf ah0 = frag_bf16(h_bf + (size_t)rowA * 64 + half * 16);
  v16bf ah1 = frag_bf16(h_bf + (size_t)rowA * 64 + 32 + half * 16);
  v8f ru[8], cx[4], ch[4];
#pragma unroll
  for (int nt = 0; nt < 8; ++nt) {
    v8f c = {};
    c = wmma_bf16(ax0, frag_bf16(WxS + ((size_t)(0 * 12 + nt) * 32 + lane) * 16), c);
    c = wmma_bf16(ax1, frag_bf16(WxS + ((size_t)(1 * 12 + nt) * 32 + lane) * 16), c);
    c = wmma_bf16(ax2, frag_bf16(WxS + ((size_t)(2 * 12 + nt) * 32 + lane) * 16), c);
    c = wmma_bf16(ax3, frag_bf16(WxS + ((size_t)(3 * 12 + nt) * 32 + lane) * 16), c);
    c = wmma_bf16(ah0, frag_bf16(WhS + ((size_t)(0 * 12 + nt) * 32 + lane) * 16), c);
    c = wmma_bf16(ah1, frag_bf16(WhS + ((size_t)(1 * 12 + nt) * 32 + lane) * 16), c);
    ru[nt] = c;
  }
#pragma unroll
  for (int nt = 0; nt < 4; ++nt) {
    v8f c = {};
    c = wmma_bf16(ax0, frag_bf16(WxS + ((size_t)(0 * 12 + 8 + nt) * 32 + lane) * 16), c);
    c = wmma_bf16(ax1, frag_bf16(WxS + ((size_t)(1 * 12 + 8 + nt) * 32 + lane) * 16), c);
    c = wmma_bf16(ax2, frag_bf16(WxS + ((size_t)(2 * 12 + 8 + nt) * 32 + lane) * 16), c);
    c = wmma_bf16(ax3, frag_bf16(WxS + ((size_t)(3 * 12 + 8 + nt) * 32 + lane) * 16), c);
    cx[nt] = c;
    v8f d = {};
    d = wmma_bf16(ah0, frag_bf16(WhS + ((size_t)(0 * 12 + 8 + nt) * 32 + lane) * 16), d);
    d = wmma_bf16(ah1, frag_bf16(WhS + ((size_t)(1 * 12 + 8 + nt) * 32 + lane) * 16), d);
    ch[nt] = d;
  }
  if (wid * 16 + 16 <= N)
    gru_epilogue<true, false>(wid, m, half, N, ru, cx, ch, bias, h, outF, nullptr);
  else
    gru_epilogue<false, false>(wid, m, half, N, ru, cx, ch, bias, h, outF, nullptr);
}

// ---------- host ----------
extern "C" void kernel_launch(void* const* d_in, const int* in_sizes, int n_in,
                              void* d_out, int out_size, void* d_ws, size_t ws_size,
                              hipStream_t stream) {
  (void)n_in; (void)out_size; (void)ws_size;
  const int* node_subtokens = (const int*)d_in[0];
  const int* adjacency      = (const int*)d_in[1];
  const float* embed_table  = (const float*)d_in[2];
  const float* proj_W       = (const float*)d_in[3];
  const float* edge_W0      = (const float*)d_in[4];
  const float* edge_W1      = (const float*)d_in[5];
  const float* gru0_Wx      = (const float*)d_in[6];
  const float* gru0_Wh      = (const float*)d_in[7];
  const float* gru0_b       = (const float*)d_in[8];
  const float* gru1_Wx      = (const float*)d_in[9];
  const float* gru1_Wh      = (const float*)d_in[10];
  const float* gru1_b       = (const float*)d_in[11];

  const int S = 5, T = 4;
  int N = in_sizes[0] / S;
  int E = in_sizes[1] / (T * 2);
  size_t nd = (size_t)N * 64;

  char* ws = (char*)d_ws;
  size_t off = 0;
  auto alloc = [&](size_t bytes) -> void* {
    void* p = ws + off;
    off = (off + bytes + 255) & ~(size_t)255;
    return p;
  };
  float*    h        = (float*)alloc(nd * 4);
  ushort_t* hbf      = (ushort_t*)alloc(nd * 2);
  ushort_t* hinitbf  = (ushort_t*)alloc(nd * 2);
  uint_t*   msgK     = (uint_t*)alloc(nd * 4);
  float*    emb      = (float*)alloc(nd * 4);
  ushort_t* swzE0    = (ushort_t*)alloc((size_t)8 * 4096 * 2);
  ushort_t* swzE1    = (ushort_t*)alloc((size_t)8 * 4096 * 2);
  ushort_t* swzWx0   = (ushort_t*)alloc((size_t)2 * 12 * 512 * 2);
  ushort_t* swzWh0   = (ushort_t*)alloc((size_t)2 * 12 * 512 * 2);
  ushort_t* swzWx1   = (ushort_t*)alloc((size_t)4 * 12 * 512 * 2);
  ushort_t* swzWh1   = (ushort_t*)alloc((size_t)2 * 12 * 512 * 2);
  ushort_t* swzP     = (ushort_t*)alloc((size_t)2 * 4 * 512 * 2);

  auto cdiv = [](int a, int b) { return (a + b - 1) / b; };
  auto sw = [&](const float* W, ushort_t* dst, int K, int NC, int batch) {
    int tot = batch * K * NC;
    swizzle_w_kernel<<<cdiv(tot, 256), 256, 0, stream>>>(W, dst, K, NC, tot);
  };
  sw(edge_W0, swzE0, 64, 64, 8);
  sw(edge_W1, swzE1, 64, 64, 8);
  sw(gru0_Wx, swzWx0, 64, 192, 1);
  sw(gru0_Wh, swzWh0, 64, 192, 1);
  sw(gru1_Wx, swzWx1, 128, 192, 1);
  sw(gru1_Wh, swzWh1, 64, 192, 1);
  sw(proj_W, swzP, 64, 64, 1);

  embed_kernel<<<cdiv(N * 64, 256), 256, 0, stream>>>(node_subtokens, embed_table, emb, N);

  int numTiles = cdiv(N, 16);
  int gruGrid = cdiv(numTiles * 32, 256);
  init_h_kernel<<<gruGrid, 256, 0, stream>>>(emb, swzP, h, hbf, hinitbf, N);

  int tilesPerType = cdiv(E, 16);
  int wavesPerType = cdiv(tilesPerType, AGG_TPW);
  int aggGrid = cdiv(2 * T * wavesPerType * 32, 256);
  int fillGrid = cdiv((int)nd, 256);

  for (int step = 0; step < 7; ++step) {
    fill_key_kernel<<<fillGrid, 256, 0, stream>>>(msgK, (int)nd);
    aggregate_kernel<<<aggGrid, 256, 0, stream>>>(adjacency, hbf, swzE0, msgK, E, 2 * T, T);
    gru0_kernel<<<gruGrid, 256, 0, stream>>>(msgK, h, hbf, swzWx0, swzWh0, gru0_b, N);
  }

  fill_key_kernel<<<fillGrid, 256, 0, stream>>>(msgK, (int)nd);
  aggregate_kernel<<<aggGrid, 256, 0, stream>>>(adjacency, hbf, swzE1, msgK, E, 2 * T, T);
  gru1_kernel<<<gruGrid, 256, 0, stream>>>(msgK, hinitbf, h, hbf, swzWx1, swzWh1, gru1_b,
                                           (float*)d_out, N);
}
